// RNN_54958401520105
// MI455X (gfx1250) — matrix operand order repaired and checked by reading, not compiled, not run on hardware
//
#include <hip/hip_runtime.h>
#include <hip/hip_bf16.h>

// CDNA5 / gfx1250: wave32, WMMA 16x16x32 bf16 -> f32 accumulate.
typedef __attribute__((ext_vector_type(16))) __bf16 v16bf;
typedef __attribute__((ext_vector_type(8)))  float  v8f;

#define B_TOT  32768
#define IN_F   256
#define NA_    49
#define NH_    64
#define L_     15
#define AXP    52      // Ax row padding (cols 49..51 zero)

// fragment buffer sizes (elements, bf16)
#define XFRAG_ELEMS (2048u * 8u * 32u * 16u)      // [mtile][kb][lane][16]      = 8,388,608
#define AFRAG_ELEMS (15u * 8u * 4u * 32u * 16u)   // [l][kb][ntile][lane][16]   =   245,760
#define GFRAG_ELEMS (15u * 98u * 4u * 32u * 16u)  // [l][kb][ntile][lane][16]   = 3,010,560
#define AXG_ELEMS   (15u * 32768u * (unsigned)AXP)               // [l][row][52]

// ---------------------------------------------------------------------------
// Prep kernels: pack f32 inputs into ISA WMMA fragment layouts (bf16).
// A-operand 16x32 bf16 lane layout: lanes 0-15 hold K=0..7,16..23 (pairs),
// lanes 16-31 hold K=8..15,24..31.  B-operand 32x16: same K order as A per lane half,
// column n = lane&15.  (CDNA5 ISA §7.12.2)
// ---------------------------------------------------------------------------

__global__ void prep_xfrag(const float* __restrict__ x, __bf16* __restrict__ xf) {
    unsigned tid  = blockIdx.x * blockDim.x + threadIdx.x;
    unsigned t    = tid & 15u;
    unsigned lane = (tid >> 4) & 31u;
    unsigned kb   = (tid >> 9) & 7u;
    unsigned mt   = tid >> 12;                       // 0..2047
    unsigned m    = mt * 16u + (lane & 15u);
    unsigned kk   = ((t < 8u) ? t : t + 8u) + ((lane >> 4) << 3);
    unsigned K    = kb * 32u + kk;
    xf[tid] = (__bf16)x[(size_t)m * IN_F + K];
}

__global__ void prep_afrag(const float* __restrict__ A, __bf16* __restrict__ af) {
    unsigned tid  = blockIdx.x * blockDim.x + threadIdx.x;
    unsigned t    = tid & 15u;
    unsigned lane = (tid >> 4) & 31u;
    unsigned nt   = (tid >> 9) & 3u;
    unsigned kb   = (tid >> 11) & 7u;
    unsigned l    = tid >> 14;                       // 0..14
    unsigned n    = nt * 16u + (lane & 15u);
    unsigned kk   = ((t < 8u) ? t : t + 8u) + ((lane >> 4) << 3); // table k order (matches A)
    unsigned K    = kb * 32u + kk;                   // feature index 0..255
    float v = (n < NA_) ? A[((size_t)l * IN_F + K) * NA_ + n] : 0.0f; // zero-pad cols 49..63
    af[tid] = (__bf16)v;
}

__global__ void prep_gfrag(const float* __restrict__ G, __bf16* __restrict__ gf) {
    unsigned tid  = blockIdx.x * blockDim.x + threadIdx.x;
    unsigned t    = tid & 15u;
    unsigned lane = (tid >> 4) & 31u;
    unsigned nt   = (tid >> 9) & 3u;
    unsigned rest = tid >> 11;
    unsigned kb   = rest % 98u;                      // K-block: i = kb/2, j0 = (kb&1)*32
    unsigned l    = rest / 98u;
    unsigned n    = nt * 16u + (lane & 15u);
    unsigned i    = kb >> 1;
    unsigned j    = ((kb & 1u) << 5) + ((t < 8u) ? t : t + 8u) + ((lane >> 4) << 3);
    gf[tid] = (__bf16)G[(((size_t)l * NA_ + i) * NH_ + j) * NH_ + n];
}

// ---------------------------------------------------------------------------
// Ax GEMM kernel: Ax[l,b,i] = sum_f x[b,f] * A_l[f,i]   (WMMA, bf16 out)
// One wave per (l, mtile): [16,256] x [256,64] with zero-padded cols.
// ---------------------------------------------------------------------------

__launch_bounds__(128)
__global__ void ax_gemm(const __bf16* __restrict__ xfrag,
                        const __bf16* __restrict__ afrag,
                        __bf16* __restrict__ axg) {
    const int lane   = threadIdx.x & 31;
    const int wave   = threadIdx.x >> 5;
    const int lane15 = lane & 15;
    const int laneHi = lane >> 4;
    const int wid    = blockIdx.x * 4 + wave;        // 0 .. 15*2048-1
    const int l      = wid >> 11;
    const int mtile  = wid & 2047;

    v8f acc[4];
    #pragma unroll
    for (int nt = 0; nt < 4; ++nt)
        #pragma unroll
        for (int q = 0; q < 8; ++q) acc[nt][q] = 0.0f;

    #pragma unroll
    for (int kb = 0; kb < 8; ++kb) {
        v16bf xa = *(const v16bf*)&xfrag[(((size_t)mtile * 8 + kb) * 32 + lane) * 16];
        #pragma unroll
        for (int nt = 0; nt < 4; ++nt) {
            v16bf ab = *(const v16bf*)&afrag[((((size_t)l * 8 + kb) * 4 + nt) * 32 + lane) * 16];
            acc[nt] = __builtin_amdgcn_wmma_f32_16x16x32_bf16(
                false, xa, false, ab, (short)0, acc[nt], false, false);
        }
    }
    // store: VGPR r -> row (r + laneHi*8), col = nt*16 + lane15 (cols 49..51 = 0)
    #pragma unroll
    for (int nt = 0; nt < 4; ++nt) {
        const int col = nt * 16 + lane15;
        if (col < AXP) {
            #pragma unroll
            for (int r = 0; r < 8; ++r) {
                const int row = mtile * 16 + laneHi * 8 + r;
                axg[((size_t)l * B_TOT + row) * AXP + col] = (__bf16)acc[nt][r];
            }
        }
    }
}

// ---------------------------------------------------------------------------
// Main fused kernel: 15 chained layers, h state resident in LDS (f32),
// per-layer bf16 h fragments in registers.  Block = 128 threads (4 waves);
// wave w exclusively owns rows [w*32, w*32+32) -> no barriers needed.
// ---------------------------------------------------------------------------

__launch_bounds__(128)
__global__ void grnn_chain(const __bf16* __restrict__ gfrag,
                           const __bf16* __restrict__ axg,
                           const float*  __restrict__ hidden0,
                           float* __restrict__ out) {
    __shared__ float h_lds[128][64];    // 32 KB: hidden state, f32

    const int lane   = threadIdx.x & 31;
    const int wave   = threadIdx.x >> 5;
    const int lane15 = lane & 15;
    const int laneHi = lane >> 4;       // 0 / 1
    const int wrow   = wave * 32;       // wave's local row base

    // per-wave init of own rows from input hidden (reference: ones)
    const size_t obase = (size_t)blockIdx.x * 128 * NH_;
    float* hflat = &h_lds[wrow][0];
    for (int idx = lane; idx < 32 * NH_; idx += 32)
        hflat[idx] = hidden0[obase + (size_t)wrow * NH_ + idx];

    const int row0g = blockIdx.x * 128 + wrow + lane15;   // M-tile 0 row
    const int row1g = row0g + 16;                          // M-tile 1 row

    for (int l = 0; l < L_; ++l) {
        // h is layer-invariant: bf16 register fragments, A-operand ordering.
        // hbf[mt][jh][t]: t=0..7 -> h[jh*32+laneHi*8+t], t=8..15 -> +16
        v16bf hbf[2][2];
        #pragma unroll
        for (int mt = 0; mt < 2; ++mt) {
            const float* hb = &h_lds[wrow + mt * 16 + lane15][laneHi * 8];
            #pragma unroll
            for (int jh = 0; jh < 2; ++jh) {
                #pragma unroll
                for (int q = 0; q < 8; ++q) hbf[mt][jh][q]     = (__bf16)hb[jh * 32 + q];
                #pragma unroll
                for (int q = 0; q < 8; ++q) hbf[mt][jh][8 + q] = (__bf16)hb[jh * 32 + 16 + q];
            }
        }

        v8f accD[2][4];
        #pragma unroll
        for (int mt = 0; mt < 2; ++mt)
            #pragma unroll
            for (int nt = 0; nt < 4; ++nt)
                #pragma unroll
                for (int q = 0; q < 8; ++q) accD[mt][nt][q] = 0.0f;

        const __bf16* gbase = &gfrag[(size_t)l * 98 * 4 * 32 * 16];
        const __bf16* axb0  = &axg[((size_t)l * B_TOT + row0g) * AXP];
        const __bf16* axb1  = &axg[((size_t)l * B_TOT + row1g) * AXP];

        // double-buffered B fragments: bfA = even kb, bfB = odd kb
        v16bf bfA[4], bfB[4];
        #pragma unroll
        for (int nt = 0; nt < 4; ++nt)
            bfA[nt] = *(const v16bf*)&gbase[((size_t)nt * 32 + lane) * 16];

        for (int i = 0; i < NA_; ++i) {
            const __bf16 ax0 = axb0[i];
            const __bf16 ax1 = axb1[i];

            // pull upcoming G fragments toward the WGP (global_prefetch_b8)
            if (i < NA_ - 2)
                __builtin_prefetch(&gbase[(((size_t)(2 * i + 4) * 4) * 32 + lane) * 16], 0, 1);

            // ---- kb = 2i (j in [0,32)): compute with bfA, prefetch bfB=kb+1
            {
                const int kb = 2 * i;
                #pragma unroll
                for (int nt = 0; nt < 4; ++nt)
                    bfB[nt] = *(const v16bf*)&gbase[(((size_t)(kb + 1) * 4 + nt) * 32 + lane) * 16];
                // A-fragments: packed bf16 vector * scalar (v_pk_mul_bf16)
                v16bf af0 = hbf[0][0] * ax0;
                v16bf af1 = hbf[1][0] * ax1;
                #pragma unroll
                for (int nt = 0; nt < 4; ++nt)
                    accD[0][nt] = __builtin_amdgcn_wmma_f32_16x16x32_bf16(
                        false, af0, false, bfA[nt], (short)0, accD[0][nt], false, false);
                #pragma unroll
                for (int nt = 0; nt < 4; ++nt)
                    accD[1][nt] = __builtin_amdgcn_wmma_f32_16x16x32_bf16(
                        false, af1, false, bfA[nt], (short)0, accD[1][nt], false, false);
            }
            // ---- kb = 2i+1 (j in [32,64)): compute with bfB, prefetch bfA
            {
                const int kb = 2 * i + 1;
                if (i < NA_ - 1) {
                    #pragma unroll
                    for (int nt = 0; nt < 4; ++nt)
                        bfA[nt] = *(const v16bf*)&gbase[(((size_t)(kb + 1) * 4 + nt) * 32 + lane) * 16];
                }
                v16bf af0 = hbf[0][1] * ax0;
                v16bf af1 = hbf[1][1] * ax1;
                #pragma unroll
                for (int nt = 0; nt < 4; ++nt)
                    accD[0][nt] = __builtin_amdgcn_wmma_f32_16x16x32_bf16(
                        false, af0, false, bfB[nt], (short)0, accD[0][nt], false, false);
                #pragma unroll
                for (int nt = 0; nt < 4; ++nt)
                    accD[1][nt] = __builtin_amdgcn_wmma_f32_16x16x32_bf16(
                        false, af1, false, bfB[nt], (short)0, accD[1][nt], false, false);
            }
        }

        // write back h' (wave-private rows; same-wave LDS ordering suffices)
        #pragma unroll
        for (int mt = 0; mt < 2; ++mt)
            #pragma unroll
            for (int nt = 0; nt < 4; ++nt)
                #pragma unroll
                for (int r = 0; r < 8; ++r)
                    h_lds[wrow + mt * 16 + laneHi * 8 + r][nt * 16 + lane15] = accD[mt][nt][r];
    }

    // final h -> out (f32, row-major [B,64]); LDS layout matches
    for (int idx = lane; idx < 32 * NH_; idx += 32)
        out[obase + (size_t)wrow * NH_ + idx] = hflat[idx];
}

// ---------------------------------------------------------------------------

extern "C" void kernel_launch(void* const* d_in, const int* in_sizes, int n_in,
                              void* d_out, int out_size, void* d_ws, size_t ws_size,
                              hipStream_t stream) {
    const float* x  = (const float*)d_in[0];   // [32768,256]
    const float* h0 = (const float*)d_in[1];   // [32768,64]
    const float* A  = (const float*)d_in[2];   // [15,256,49]
    const float* G  = (const float*)d_in[3];   // [15,49,64,64]
    float* out = (float*)d_out;

    __bf16* xf  = (__bf16*)d_ws;
    __bf16* af  = xf + XFRAG_ELEMS;
    __bf16* gf  = af + AFRAG_ELEMS;
    __bf16* axg = gf + GFRAG_ELEMS;

    prep_xfrag<<<XFRAG_ELEMS / 256, 256, 0, stream>>>(x, xf);
    prep_afrag<<<AFRAG_ELEMS / 256, 256, 0, stream>>>(A, af);
    prep_gfrag<<<GFRAG_ELEMS / 256, 256, 0, stream>>>(G, gf);

    // Ax for all 15 layers: one wave per (l, mtile) -> 15*2048/4 blocks
    ax_gemm<<<(15 * 2048) / 4, 128, 0, stream>>>(xf, af, axg);

    grnn_chain<<<B_TOT / 128, 128, 0, stream>>>(gf, axg, h0, out);
}
